// STDGAT_9208409883388
// MI455X (gfx1250) — matrix operand order repaired and checked by reading, not compiled, not run on hardware
//
#include <hip/hip_runtime.h>
#include <hip/hip_bf16.h>
#include <math.h>

// ---- problem constants (from reference) ----
#define SS   24      // sequence length
#define BB   128     // batch
#define NN   121     // graph nodes
#define FF   32      // GAT feature width
#define DD   (NN*FF) // 3872, LSTM input width
#define HH   512     // LSTM hidden
#define G4   (4*HH)  // 2048 gates
#define MX   (SS*BB) // 3072 rows of hoisted input GEMM

typedef __attribute__((ext_vector_type(16))) _Float16 v16h;
typedef __attribute__((ext_vector_type(8)))  _Float16 h8;
typedef __attribute__((ext_vector_type(8)))  float    v8f;

static __device__ __forceinline__ v16h cat16(h8 lo, h8 hi) {
  return __builtin_shufflevector(lo, hi, 0,1,2,3,4,5,6,7,8,9,10,11,12,13,14,15);
}
static __device__ __forceinline__ float lrelu(float v) { return v > 0.f ? v : 0.01f * v; }
static __device__ __forceinline__ float sigm(float v)  { return 1.f / (1.f + __expf(-v)); }

// ---------------- f32 -> f16 weight conversion ----------------
__global__ void f32_to_f16_kernel(const float* __restrict__ in, _Float16* __restrict__ out, int n) {
  int i = blockIdx.x * blockDim.x + threadIdx.x;
  if (i < n) out[i] = (_Float16)in[i];
}

// ---------------- bias = b_ih + b_hh ----------------
__global__ void bias_kernel(const float* __restrict__ a, const float* __restrict__ b,
                            float* __restrict__ o, int n) {
  int i = blockIdx.x * blockDim.x + threadIdx.x;
  if (i < n) o[i] = a[i] + b[i];
}

// ---------------- fused 3-layer GAT chain ----------------
// softmax rows sum to 1 => each layer is exactly leaky_relu(h @ W).
__global__ void gat_kernel(const float* __restrict__ x, const float* __restrict__ W1,
                           const float* __restrict__ W2, const float* __restrict__ W3,
                           _Float16* __restrict__ X16) {
  __shared__ float sW1[FF], sW2[FF*FF], sW3[FF*FF];
  for (int i = threadIdx.x; i < FF*FF; i += blockDim.x) { sW2[i] = W2[i]; sW3[i] = W3[i]; }
  if (threadIdx.x < FF) sW1[threadIdx.x] = W1[threadIdx.x];
  __syncthreads();

  int id = blockIdx.x * blockDim.x + threadIdx.x;
  if (id >= SS*BB*NN) return;
  int n  = id % NN;
  int sb = id / NN;
  int b  = sb % BB;
  int s  = sb / BB;

  // x layout (B,S,N,1); reference swaps to (S,B,...)
  float xv = x[((size_t)b*SS + s)*NN + n];

  float h1[FF], h2[FF];
  #pragma unroll
  for (int f = 0; f < FF; ++f) h1[f] = lrelu(xv * sW1[f]);
  #pragma unroll 4
  for (int f = 0; f < FF; ++f) {
    float acc = 0.f;
    #pragma unroll
    for (int k = 0; k < FF; ++k) acc += h1[k] * sW2[k*FF + f];
    h2[f] = lrelu(acc);
  }
  _Float16* dst = X16 + ((size_t)(s*BB + b))*DD + (size_t)n*FF;
  #pragma unroll 4
  for (int f = 0; f < FF; ++f) {
    float acc = 0.f;
    #pragma unroll
    for (int k = 0; k < FF; ++k) acc += h2[k] * sW3[k*FF + f];
    dst[f] = (_Float16)lrelu(acc);
  }
}

// ---------------- WMMA GEMM: C[M,N] = A[M,K] * Bt[N,K]^T (+Cadd)(+bias) ----------------
// A row-major f16, Bt row-major (N,K) f16 (i.e. B[k][n] = Bt[n*K+k]).
// One wave computes a 16 x (16*NT) strip: NT accumulators share each A fragment.
// Software-pipelined: iteration ks issues loads for ks+1, then runs NT WMMAs on
// the fragments loaded last iteration, so load latency overlaps the matrix pipe.
// K must be a multiple of 32; M of 16; N of 16*NT.
template <int NT>
__global__ void wmma_gemm_nt(const _Float16* __restrict__ A, const _Float16* __restrict__ Bt,
                             const float* __restrict__ Cadd, const float* __restrict__ bias,
                             float* __restrict__ C, int M, int N, int K) {
  int wave = threadIdx.x >> 5;
  int lane = threadIdx.x & 31;
  int gid  = blockIdx.x * (blockDim.x >> 5) + wave;
  int gn   = N / (16 * NT);          // strip groups along N
  int m0   = (gid / gn) << 4;
  int nb   = (gid % gn) * (16 * NT); // first column of this wave's strip
  if (m0 >= M) return;
  int half = lane >> 4;   // 0 or 1
  int l16  = lane & 15;

#if defined(__gfx1250__)
  // ISA layouts (7.12.2): A lane l -> row l16, K slices [half*8,+8) and [16+half*8,+8)
  //                       B lane l -> col l16, K slices [half*8,+8) and [16+half*8,+8)
  const _Float16* arow = A + (size_t)(m0 + l16) * K + half*8;
  const _Float16* brow[NT];
  #pragma unroll
  for (int t = 0; t < NT; ++t)
    brow[t] = Bt + (size_t)(nb + t*16 + l16) * K + half*8;

  v8f acc[NT];
  #pragma unroll
  for (int t = 0; t < NT; ++t) acc[t] = (v8f){};

  // prologue: fragments for k0 = 0
  v16h a_cur = cat16(*(const h8*)(arow), *(const h8*)(arow + 16));
  v16h b_cur[NT];
  #pragma unroll
  for (int t = 0; t < NT; ++t)
    b_cur[t] = cat16(*(const h8*)(brow[t]), *(const h8*)(brow[t] + 16));

  const int ksteps = K >> 5;
  for (int ks = 0; ks < ksteps - 1; ++ks) {
    int kn = (ks + 1) << 5;
    // issue next-iteration loads first (no consumer in this iteration)
    v16h a_nxt = cat16(*(const h8*)(arow + kn), *(const h8*)(arow + kn + 16));
    v16h b_nxt[NT];
    #pragma unroll
    for (int t = 0; t < NT; ++t)
      b_nxt[t] = cat16(*(const h8*)(brow[t] + kn), *(const h8*)(brow[t] + kn + 16));
    // compute on current fragments while loads are in flight
    #pragma unroll
    for (int t = 0; t < NT; ++t)
      acc[t] = __builtin_amdgcn_wmma_f32_16x16x32_f16(
          false, a_cur, false, b_cur[t], (short)0, acc[t], false, false);
    a_cur = a_nxt;
    #pragma unroll
    for (int t = 0; t < NT; ++t) b_cur[t] = b_nxt[t];
  }
  // epilogue: last K-step
  #pragma unroll
  for (int t = 0; t < NT; ++t)
    acc[t] = __builtin_amdgcn_wmma_f32_16x16x32_f16(
        false, a_cur, false, b_cur[t], (short)0, acc[t], false, false);

  // C layout: v8f element r -> row m0 + half*8 + r, col strip base + l16
  #pragma unroll
  for (int t = 0; t < NT; ++t) {
    int col = nb + t*16 + l16;
    float bv = bias ? bias[col] : 0.f;
    #pragma unroll
    for (int r = 0; r < 8; ++r) {
      size_t idx = (size_t)(m0 + half*8 + r) * N + col;
      float v = acc[t][r] + bv;
      if (Cadd) v += Cadd[idx];
      C[idx] = v;
    }
  }
#else
  // semantic fallback for host pass (never executed on gfx1250)
  if (lane == 0) {
    for (int i = 0; i < 16; ++i)
      for (int j = 0; j < 16*NT; ++j) {
        float s = bias ? bias[nb + j] : 0.f;
        for (int k = 0; k < K; ++k)
          s += (float)A[(size_t)(m0+i)*K + k] * (float)Bt[(size_t)(nb+j)*K + k];
        size_t idx = (size_t)(m0+i)*N + nb + j;
        if (Cadd) s += Cadd[idx];
        C[idx] = s;
      }
  }
#endif
}

// ---------------- init h/c state ----------------
__global__ void init_state_kernel(float* __restrict__ c, float* __restrict__ h,
                                  _Float16* __restrict__ h16) {
  int i = blockIdx.x * blockDim.x + threadIdx.x;
  if (i < BB*HH) { c[i] = 0.f; h[i] = 0.f; h16[i] = (_Float16)0.f; }
}

// ---------------- LSTM pointwise update ----------------
__global__ void lstm_point_kernel(const float* __restrict__ gates, float* __restrict__ c,
                                  float* __restrict__ h, _Float16* __restrict__ h16) {
  int t = blockIdx.x * blockDim.x + threadIdx.x;
  if (t >= BB*HH) return;
  int b = t >> 9;        // /512
  int j = t & (HH - 1);
  const float* g = gates + (size_t)b * G4;
  float iv = sigm(g[j]);
  float fv = sigm(g[HH + j]);
  float gv = tanhf(g[2*HH + j]);
  float ov = sigm(g[3*HH + j]);
  float cn = fv * c[t] + iv * gv;
  float hn = ov * tanhf(cn);
  c[t] = cn;
  h[t] = hn;
  h16[t] = (_Float16)hn;
}

// ---------------- final FC + ReLU ----------------
__global__ void fc_kernel(const float* __restrict__ h, const float* __restrict__ fcW,
                          const float* __restrict__ fcb, float* __restrict__ out) {
  int b = blockIdx.x;
  int n = threadIdx.x;
  if (n >= NN) return;
  const float* hr = h + (size_t)b * HH;
  const float* wr = fcW + (size_t)n * HH;
  float acc = fcb[n];
  #pragma unroll 8
  for (int k = 0; k < HH; ++k) acc += hr[k] * wr[k];
  out[(size_t)b*NN + n] = acc > 0.f ? acc : 0.f;
}

extern "C" void kernel_launch(void* const* d_in, const int* in_sizes, int n_in,
                              void* d_out, int out_size, void* d_ws, size_t ws_size,
                              hipStream_t stream) {
  (void)in_sizes; (void)n_in; (void)out_size; (void)ws_size;
  const float* x     = (const float*)d_in[0];
  // d_in[1] (features/adjacency), d_in[3,5,7] (a1,a2,a3) are mathematically dead:
  // sum(softmax(e), axis) == 1, so each GAT layer reduces to leaky_relu(h @ W).
  const float* W1    = (const float*)d_in[2];
  const float* W2    = (const float*)d_in[4];
  const float* W3    = (const float*)d_in[6];
  const float* W_ih  = (const float*)d_in[8];
  const float* W_hh  = (const float*)d_in[9];
  const float* b_ih  = (const float*)d_in[10];
  const float* b_hh  = (const float*)d_in[11];
  const float* fcW   = (const float*)d_in[12];
  const float* fcb   = (const float*)d_in[13];
  float* out = (float*)d_out;

  // workspace carve-out (256B aligned)
  char* p = (char*)d_ws;
  auto carve = [&](size_t bytes) -> char* {
    char* r = p; p += (bytes + 255) & ~(size_t)255; return r;
  };
  _Float16* X16   = (_Float16*)carve((size_t)MX * DD * 2);   // 23.8 MB GAT output / LSTM input
  _Float16* Wih16 = (_Float16*)carve((size_t)G4 * DD * 2);   // 15.9 MB
  _Float16* Whh16 = (_Float16*)carve((size_t)G4 * HH * 2);   //  2.0 MB
  float*    bias  = (float*)   carve((size_t)G4 * 4);
  float*    XW    = (float*)   carve((size_t)MX * G4 * 4);   // 25.2 MB hoisted x@W_ih^T + bias
  float*    gates = (float*)   carve((size_t)BB * G4 * 4);   //  1.0 MB
  float*    cst   = (float*)   carve((size_t)BB * HH * 4);
  float*    hst   = (float*)   carve((size_t)BB * HH * 4);
  _Float16* h16   = (_Float16*)carve((size_t)BB * HH * 2);

  // 1) weight conversions + fused bias
  {
    int n = G4 * DD;
    f32_to_f16_kernel<<<(n + 255)/256, 256, 0, stream>>>(W_ih, Wih16, n);
  }
  {
    int n = G4 * HH;
    f32_to_f16_kernel<<<(n + 255)/256, 256, 0, stream>>>(W_hh, Whh16, n);
  }
  bias_kernel<<<(G4 + 255)/256, 256, 0, stream>>>(b_ih, b_hh, bias, G4);

  // 2) fused GAT chain -> f16 LSTM input
  gat_kernel<<<(SS*BB*NN + 127)/128, 128, 0, stream>>>(x, W1, W2, W3, X16);

  // 3) hoisted input projection for all timesteps: XW = X16 @ W_ih^T + bias
  //    waves = (3072/16)*(2048/64) = 6144 strips; 8 waves per 256-thread block
  wmma_gemm_nt<4><<<(MX/16)*(G4/64)/8, 256, 0, stream>>>(X16, Wih16, nullptr, bias, XW, MX, G4, DD);

  // 4) LSTM scan
  init_state_kernel<<<(BB*HH + 255)/256, 256, 0, stream>>>(cst, hst, h16);
  for (int s = 0; s < SS; ++s) {
    // gates = h @ W_hh^T + XW[s] -> waves = (128/16)*(2048/32) = 512 strips -> 64 blocks
    wmma_gemm_nt<2><<<(BB/16)*(G4/32)/8, 256, 0, stream>>>(
        h16, Whh16, XW + (size_t)s * BB * G4, nullptr, gates, BB, G4, HH);
    lstm_point_kernel<<<(BB*HH + 255)/256, 256, 0, stream>>>(gates, cst, hst, h16);
  }

  // 5) out = relu(h_last @ fc_W^T + fc_b)
  fc_kernel<<<BB, 128, 0, stream>>>(hst, fcW, fcb, out);
}
